// TacoMamba_4303557231106
// MI455X (gfx1250) — matrix operand order repaired and checked by reading, not compiled, not run on hardware
//
#include <hip/hip_runtime.h>
#include <math.h>

// ---------------------------------------------------------------------------
// TacoMamba forward for gfx1250 (MI455X, wave32, WMMA).
// GEMMs run on v_wmma_f32_16x16x32_f16 (f32 in -> f16 tiles in LDS -> f32 acc).
// All GEMM shapes in this net are multiples of the tile sizes (M%64==0,
// N%16==0, K%32==0) so staging is branch-free and fully vectorized.
// Wave tile is 16x32 (two WMMAs sharing one A fragment) when N%32==0.
// ---------------------------------------------------------------------------

typedef __attribute__((ext_vector_type(16))) _Float16 v16h;
typedef __attribute__((ext_vector_type(8)))  _Float16 h8;
typedef __attribute__((ext_vector_type(4)))  _Float16 h4;
typedef __attribute__((ext_vector_type(8)))  float    v8f;

#define BATCH   4
#define T_ENC   256
#define T_DEC   512
#define DMODEL  512
#define DINNER  1024
#define DSTATE  16
#define DTRANK  32
#define NMELS   80
#define NLAYER  3
#define BM      64      // rows per block (4 waves x 16)

__device__ __forceinline__ float siluf(float x)     { return x / (1.f + __expf(-x)); }
__device__ __forceinline__ float softplusf(float x) { return (x > 20.f) ? x : log1pf(__expf(x)); }

enum { EPI_NONE = 0, EPI_BIAS_SOFTPLUS = 1, EPI_RELU_BN = 2, EPI_ROWSCALE = 3 };

// ---------------------------------------------------------------------------
// WMMA GEMM: C[M,N] = A[M,K] @ B[K,N] (row-major f32). Block = 128 threads
// (4 waves); block tile 64 x (16*NT), wave tile 16 x (16*NT), K stepped by 32.
// Requires: M%64==0, N%(16*NT)==0, K%32==0, lda%4==0.
// Fragment layouts per CDNA5 ISA 7.12.2.
// ---------------------------------------------------------------------------
template <int NT>
__global__ void __launch_bounds__(128)
wmma_gemm_kernel(const float* __restrict__ A, int lda,
                 const float* __restrict__ Bm, int ldb,
                 float* __restrict__ C, int ldc,
                 int M, int N, int K, int epi,
                 const float* __restrict__ e0,
                 const float* __restrict__ e1,
                 const float* __restrict__ e2)
{
    __shared__ _Float16 As[BM * 32];         // [m in 0..63][k in 0..31]
    __shared__ _Float16 BsT[16 * NT * 32];   // transposed: [n][k in 0..31]
    const int tid  = threadIdx.x;            // 0..127
    const int lane = tid & 31;
    const int wv   = tid >> 5;               // wave id 0..3 -> M sub-tile
    const int tn   = blockIdx.x, tm = blockIdx.y;
    const int col  = lane & 15;              // index within 16-wide tile
    const int hl   = lane >> 4;              // lane half
    const int m0   = tm * BM;
    const int n0   = tn * (16 * NT);

    // ---- precompute staging pointers (advance by constant strides) ----
    const float* ap_g[4];
    _Float16*    as_st[4];
#pragma unroll
    for (int it = 0; it < 4; ++it) {
        int idx = tid + 128 * it;            // 0..511 quads of A tile
        int r = idx >> 3, cq = (idx & 7) * 4;
        ap_g[it]  = A + (size_t)(m0 + r) * lda + cq;
        as_st[it] = As + r * 32 + cq;
    }
    const int bcol = tid % (16 * NT);
    const int bkq  = (tid / (16 * NT)) * (4 * NT);
    const float* bp_g = Bm + (size_t)bkq * ldb + n0 + bcol;
    _Float16*    bs_st = BsT + bcol * 32 + bkq;

    // ---- fragment read pointers ----
    const _Float16* ap_f = As + (wv * 16 + col) * 32 + hl * 8;

    v8f acc[NT];
#pragma unroll
    for (int nt = 0; nt < NT; ++nt) acc[nt] = (v8f){};

    for (int k0 = 0; k0 < K; k0 += 32) {
        // ---- load phase: all global loads issued before any use ----
        float4 av[4];
#pragma unroll
        for (int it = 0; it < 4; ++it) av[it] = *(const float4*)ap_g[it];
        float bv[4 * NT];
#pragma unroll
        for (int j = 0; j < 4 * NT; ++j) bv[j] = bp_g[(size_t)j * ldb];
#pragma unroll
        for (int it = 0; it < 4; ++it) ap_g[it] += 32;
        bp_g += (size_t)32 * ldb;
        // prefetch next K chunk (pointers already advanced)
        if (k0 + 32 < K) {
            __builtin_prefetch(ap_g[0], 0, 3);
            __builtin_prefetch(bp_g, 0, 3);
        }

        // ---- convert + LDS store phase ----
#pragma unroll
        for (int it = 0; it < 4; ++it) {
            h4 hv = { (_Float16)av[it].x, (_Float16)av[it].y,
                      (_Float16)av[it].z, (_Float16)av[it].w };
            *(h4*)as_st[it] = hv;
        }
#pragma unroll
        for (int q = 0; q < NT; ++q) {
            h4 hv = { (_Float16)bv[q*4+0], (_Float16)bv[q*4+1],
                      (_Float16)bv[q*4+2], (_Float16)bv[q*4+3] };
            *(h4*)(bs_st + q * 4) = hv;
        }
        __syncthreads();

        // ---- fragments (contiguous b128 LDS reads per ISA layout) ----
        h8 a0 = *(const h8*)ap_f;            // K = hl*8 + 0..7
        h8 a1 = *(const h8*)(ap_f + 16);     // K = 16 + hl*8 + 0..7
        v16h af = __builtin_shufflevector(a0, a1, 0,1,2,3,4,5,6,7,8,9,10,11,12,13,14,15);
#pragma unroll
        for (int nt = 0; nt < NT; ++nt) {
            const _Float16* bp2 = BsT + (nt * 16 + col) * 32 + hl * 8;
            h8 b0 = *(const h8*)bp2;         // K = hl*8 + 0..7
            h8 b1 = *(const h8*)(bp2 + 16);  // K = 16 + hl*8 + 0..7
            v16h bf = __builtin_shufflevector(b0, b1, 0,1,2,3,4,5,6,7,8,9,10,11,12,13,14,15);
            acc[nt] = __builtin_amdgcn_wmma_f32_16x16x32_f16(false, af, false, bf,
                                                             (short)0, acc[nt], false, false);
        }
        __syncthreads();
    }

    const float inv_bn = rsqrtf(1.f + 1e-5f);  // BN eval: running var 1 + eps
#pragma unroll
    for (int nt = 0; nt < NT; ++nt) {
#pragma unroll
        for (int r = 0; r < 8; ++r) {
            int gm = m0 + wv * 16 + r + 8 * hl;  // C/D: VGPR r -> M=r (+8 hi half)
            int gn = n0 + nt * 16 + col;
            float v = acc[nt][r];
            if (epi == EPI_BIAS_SOFTPLUS)  v = softplusf(v + e0[gn]);
            else if (epi == EPI_RELU_BN)   v = fmaxf(v + e0[gn], 0.f) * (e1[gn] * inv_bn) + e2[gn];
            else if (epi == EPI_ROWSCALE)  v = v * e0[gm];
            C[(size_t)gm * ldc + gn] = v;
        }
    }
}

// ---------------------------------------------------------------------------
// Elementwise / pointwise kernels
// ---------------------------------------------------------------------------
__global__ void embed_kernel(const int* __restrict__ text,
                             const float* __restrict__ emb,
                             float* __restrict__ out, int total)
{
    int i = blockIdx.x * blockDim.x + threadIdx.x;
    if (i >= total) return;
    int rowd = i / DMODEL, d = i % DMODEL;
    out[i] = emb[(size_t)text[rowd] * DMODEL + d];
}

// causal depthwise conv (K=4, left pad 3) + SiLU; reads xi half of xz [B*L,2048]
__global__ void dwconv_silu_kernel(const float* __restrict__ xz,
                                   const float* __restrict__ w,
                                   const float* __restrict__ b,
                                   float* __restrict__ xi, int L, int total)
{
    int i = blockIdx.x * blockDim.x + threadIdx.x;
    if (i >= total) return;
    int d = i % DINNER;
    int rowd = i / DINNER;
    int l = rowd % L, bb = rowd / L;
    float acc = b[d];
#pragma unroll
    for (int k = 0; k < 4; ++k) {
        int ll = l - 3 + k;
        if (ll >= 0)
            acc += xz[(size_t)(bb * L + ll) * (2 * DINNER) + d] * w[d * 4 + k];
    }
    xi[i] = siluf(acc);
}

// selective scan fused with + u*D and SiLU(res) gate.
// one thread per (b, d); 16-element state in registers, sequential over L.
__global__ void scan_kernel(const float* __restrict__ delta,
                            const float* __restrict__ xi,
                            const float* __restrict__ xdbl,
                            const float* __restrict__ A_log,
                            const float* __restrict__ Dp,
                            const float* __restrict__ xz,
                            float* __restrict__ out, int L)
{
    int t = blockIdx.x * blockDim.x + threadIdx.x;
    if (t >= BATCH * DINNER) return;
    int b = t / DINNER, d = t % DINNER;
    float Ar[DSTATE], h[DSTATE];
#pragma unroll
    for (int n = 0; n < DSTATE; ++n) {
        Ar[n] = -__expf(A_log[d * DSTATE + n]);
        h[n] = 0.f;
    }
    const float Dd = Dp[d];
    for (int l = 0; l < L; ++l) {
        size_t rowd = (size_t)b * L + l;
        float dl = delta[rowd * DINNER + d];
        float u  = xi[rowd * DINNER + d];
        float du = dl * u;
        const float* Bv = xdbl + rowd * 64 + DTRANK;           // cols 32..47
        const float* Cv = xdbl + rowd * 64 + DTRANK + DSTATE;  // cols 48..63
        float acc = 0.f;
#pragma unroll
        for (int n = 0; n < DSTATE; ++n) {
            h[n] = __expf(dl * Ar[n]) * h[n] + du * Bv[n];
            acc += h[n] * Cv[n];
        }
        float res = xz[rowd * (2 * DINNER) + DINNER + d];
        out[rowd * DINNER + d] = (acc + u * Dd) * siluf(res);
    }
}

// im2col for K=3, pad=1 conv over time; X [B*L, 512] -> col [B*L, 1536] (c*3+k)
__global__ void im2col_k3_kernel(const float* __restrict__ X,
                                 float* __restrict__ col, int L, int total)
{
    int i = blockIdx.x * blockDim.x + threadIdx.x;
    if (i >= total) return;
    int k = i % 3, c = (i / 3) % DMODEL;
    int rowd = i / (3 * DMODEL);
    int l = rowd % L, b = rowd / L;
    int ll = l + k - 1;
    col[i] = (ll >= 0 && ll < L) ? X[(size_t)(b * L + ll) * DMODEL + c] : 0.f;
}

// repack conv weight [O=512][C=512][K=3] -> [(c*3+k)][o] for GEMM B operand
__global__ void wpack_k3_kernel(const float* __restrict__ w,
                                float* __restrict__ wp, int total)
{
    int i = blockIdx.x * blockDim.x + threadIdx.x;
    if (i >= total) return;
    int k = i % 3, c = (i / 3) % DMODEL, o = i / (3 * DMODEL);
    wp[(size_t)(c * 3 + k) * DMODEL + o] = w[i];
}

// 1x1 conv to scalar + softplus: dur[row] = softplus(sum_c h2[row,c]*w[c] + b)
__global__ void dur_kernel(const float* __restrict__ h2,
                           const float* __restrict__ w,
                           const float* __restrict__ b,
                           float* __restrict__ dur)
{
    int rowd = blockIdx.x;
    int lane = threadIdx.x;
    float s = 0.f;
    for (int c = lane; c < DMODEL; c += 32)
        s += h2[(size_t)rowd * DMODEL + c] * w[c];
#pragma unroll
    for (int off = 16; off; off >>= 1) s += __shfl_xor(s, off, 32);
    if (lane == 0) dur[rowd] = softplusf(s + b[0]);
}

// per-batch serial cumsum of round(clip(dur,0))
__global__ void cumsum_kernel(const float* __restrict__ dur, float* __restrict__ ends)
{
    if (threadIdx.x != 0) return;
    int b = blockIdx.x;
    float s = 0.f;
    for (int l = 0; l < T_ENC; ++l) {
        s += rintf(fmaxf(dur[b * T_ENC + l], 0.f));
        ends[b * T_ENC + l] = s;
    }
}

// searchsorted(ends, t, 'right') + mask
__global__ void expand_idx_kernel(const float* __restrict__ ends,
                                  int* __restrict__ idx, float* __restrict__ maskf)
{
    int i = blockIdx.x * blockDim.x + threadIdx.x;
    if (i >= BATCH * T_DEC) return;
    int b = i / T_DEC, t = i % T_DEC;
    const float* e = ends + b * T_ENC;
    float tf = (float)t;
    int lo = 0, hi = T_ENC;
    while (lo < hi) {
        int mid = (lo + hi) >> 1;
        if (e[mid] <= tf) lo = mid + 1; else hi = mid;
    }
    idx[i] = (lo > T_ENC - 1) ? (T_ENC - 1) : lo;
    maskf[i] = (tf < e[T_ENC - 1]) ? 1.f : 0.f;
}

__global__ void gather_kernel(const float* __restrict__ enc,
                              const int* __restrict__ idx,
                              const float* __restrict__ maskf,
                              float* __restrict__ dec, int total)
{
    int i = blockIdx.x * blockDim.x + threadIdx.x;
    if (i >= total) return;
    int d = i % DMODEL;
    int rowd = i / DMODEL;
    int b = rowd / T_DEC;
    dec[i] = enc[(size_t)(b * T_ENC + idx[rowd]) * DMODEL + d] * maskf[rowd];
}

__global__ void rmsnorm_kernel(const float* __restrict__ x,
                               const float* __restrict__ g,
                               float* __restrict__ y)
{
    int rowd = blockIdx.x;
    int lane = threadIdx.x;
    float s = 0.f;
    for (int c = lane; c < DMODEL; c += 32) {
        float v = x[(size_t)rowd * DMODEL + c];
        s += v * v;
    }
#pragma unroll
    for (int off = 16; off; off >>= 1) s += __shfl_xor(s, off, 32);
    float r = rsqrtf(s / (float)DMODEL + 1e-6f);
    for (int c = lane; c < DMODEL; c += 32)
        y[(size_t)rowd * DMODEL + c] = x[(size_t)rowd * DMODEL + c] * r * g[c];
}

__global__ void mask_out_kernel(const float* __restrict__ maskf, float* __restrict__ out)
{
    int i = blockIdx.x * blockDim.x + threadIdx.x;
    if (i < BATCH * T_DEC) out[i] = maskf[i];
}

// ---------------------------------------------------------------------------
// Host-side orchestration
// ---------------------------------------------------------------------------
static inline void gemm(hipStream_t s, const float* A, int lda, const float* B, int ldb,
                        float* C, int ldc, int M, int N, int K, int epi,
                        const float* e0 = nullptr, const float* e1 = nullptr,
                        const float* e2 = nullptr)
{
    if ((N & 31) == 0) {
        dim3 grid(N / 32, M / BM);
        wmma_gemm_kernel<2><<<grid, 128, 0, s>>>(A, lda, B, ldb, C, ldc, M, N, K, epi, e0, e1, e2);
    } else {
        dim3 grid(N / 16, M / BM);
        wmma_gemm_kernel<1><<<grid, 128, 0, s>>>(A, lda, B, ldb, C, ldc, M, N, K, epi, e0, e1, e2);
    }
}

static void mamba_layer(hipStream_t s, int L, const float* x, float* xnxt,
                        float* xz, float* xi, float* delta, float* ygate, float* xdbl,
                        const float* in_w, const float* conv_w, const float* conv_b,
                        const float* xproj_w, const float* dt_w, const float* dt_b,
                        const float* A_log, const float* Dp, const float* out_w)
{
    const int M = BATCH * L;
    // in_proj: [M,512] @ [512,2048]
    gemm(s, x, DMODEL, in_w, 2 * DINNER, xz, 2 * DINNER, M, 2 * DINNER, DMODEL, EPI_NONE);
    // causal depthwise conv + SiLU on xi half
    int n1 = M * DINNER;
    dwconv_silu_kernel<<<(n1 + 255) / 256, 256, 0, s>>>(xz, conv_w, conv_b, xi, L, n1);
    // x_proj: [M,1024] @ [1024,64]
    gemm(s, xi, DINNER, xproj_w, 64, xdbl, 64, M, 64, DINNER, EPI_NONE);
    // delta = softplus(dt @ dt_w + dt_b): A = xdbl cols 0..31 (lda=64, K=32)
    gemm(s, xdbl, 64, dt_w, DINNER, delta, DINNER, M, DINNER, DTRANK, EPI_BIAS_SOFTPLUS, dt_b);
    // selective scan + gate
    scan_kernel<<<(BATCH * DINNER + 255) / 256, 256, 0, s>>>(delta, xi, xdbl, A_log, Dp, xz, ygate, L);
    // out_proj: [M,1024] @ [1024,512]
    gemm(s, ygate, DINNER, out_w, DMODEL, xnxt, DMODEL, M, DMODEL, DINNER, EPI_NONE);
}

extern "C" void kernel_launch(void* const* d_in, const int* in_sizes, int n_in,
                              void* d_out, int out_size, void* d_ws, size_t ws_size,
                              hipStream_t stream)
{
    // ---- inputs (setup_inputs dict order) ----
    const int*   text      = (const int*)  d_in[0];
    const float* embedding = (const float*)d_in[1];
    const float* dp_c1_w   = (const float*)d_in[2];
    const float* dp_c1_b   = (const float*)d_in[3];
    const float* dp_g1     = (const float*)d_in[4];
    const float* dp_b1     = (const float*)d_in[5];
    const float* dp_c2_w   = (const float*)d_in[6];
    const float* dp_c2_b   = (const float*)d_in[7];
    const float* dp_g2     = (const float*)d_in[8];
    const float* dp_b2     = (const float*)d_in[9];
    const float* dp_c3_w   = (const float*)d_in[10];
    const float* dp_c3_b   = (const float*)d_in[11];
    const float* norm_g    = (const float*)d_in[12];
    const float* out_w_fin = (const float*)d_in[13];
    const float* enc_base[9], *dec_base[9];
    for (int i = 0; i < 9; ++i) { enc_base[i] = (const float*)d_in[14 + i];
                                  dec_base[i] = (const float*)d_in[23 + i]; }
    // per-layer strides (floats)
    const size_t ST[9] = {
        (size_t)DMODEL * 2 * DINNER,   // in_w
        (size_t)DINNER * 4,            // conv_w
        (size_t)DINNER,                // conv_b
        (size_t)DINNER * 64,           // xproj_w
        (size_t)DTRANK * DINNER,       // dt_w
        (size_t)DINNER,                // dt_b
        (size_t)DINNER * DSTATE,       // A_log
        (size_t)DINNER,                // D
        (size_t)DINNER * DMODEL        // out_w
    };

    // ---- workspace layout (floats) ----
    float* ws = (float*)d_ws;
    size_t off = 0;
    float* x_a   = ws + off; off += (size_t)BATCH * T_DEC * DMODEL;   // 1,048,576
    float* x_b   = ws + off; off += (size_t)BATCH * T_DEC * DMODEL;
    float* xz    = ws + off; off += (size_t)BATCH * T_DEC * 2 * DINNER;
    float* xi    = ws + off; off += (size_t)BATCH * T_DEC * DINNER;
    float* delta = ws + off; off += (size_t)BATCH * T_DEC * DINNER;
    float* ygate = ws + off; off += (size_t)BATCH * T_DEC * DINNER;
    float* xdbl  = ws + off; off += (size_t)BATCH * T_DEC * 64;
    float* dur   = ws + off; off += (size_t)BATCH * T_ENC;
    float* ends  = ws + off; off += (size_t)BATCH * T_ENC;
    float* maskf = ws + off; off += (size_t)BATCH * T_DEC;
    int*   idx   = (int*)(ws + off); off += (size_t)BATCH * T_DEC;
    // duration-predictor scratch overlays large buffers no longer live then:
    float* colb  = xz;     // needs B*T_ENC*1536 = 1,572,864 <= 4,194,304
    float* wpack = xi;     // needs 1536*512 = 786,432 <= 2,097,152
    float* h1    = delta;  // needs B*T_ENC*512 = 524,288
    float* h2    = ygate;

    // ---- encoder ----
    {
        int tot = BATCH * T_ENC * DMODEL;
        embed_kernel<<<(tot + 255) / 256, 256, 0, stream>>>(text, embedding, x_a, tot);
    }
    float* cur = x_a; float* nxt = x_b;
    for (int i = 0; i < NLAYER; ++i) {
        mamba_layer(stream, T_ENC, cur, nxt, xz, xi, delta, ygate, xdbl,
                    enc_base[0] + i * ST[0], enc_base[1] + i * ST[1], enc_base[2] + i * ST[2],
                    enc_base[3] + i * ST[3], enc_base[4] + i * ST[4], enc_base[5] + i * ST[5],
                    enc_base[6] + i * ST[6], enc_base[7] + i * ST[7], enc_base[8] + i * ST[8]);
        float* t = cur; cur = nxt; nxt = t;
    }
    // enc_out == cur (x_b after 3 layers)

    // ---- duration predictor ----
    {
        const int Menc = BATCH * T_ENC;
        int tcol = Menc * 3 * DMODEL;
        int twp  = DMODEL * DMODEL * 3;
        // conv1 + relu + BN
        im2col_k3_kernel<<<(tcol + 255) / 256, 256, 0, stream>>>(cur, colb, T_ENC, tcol);
        wpack_k3_kernel<<<(twp + 255) / 256, 256, 0, stream>>>(dp_c1_w, wpack, twp);
        gemm(stream, colb, 3 * DMODEL, wpack, DMODEL, h1, DMODEL,
             Menc, DMODEL, 3 * DMODEL, EPI_RELU_BN, dp_c1_b, dp_g1, dp_b1);
        // conv2 + relu + BN
        im2col_k3_kernel<<<(tcol + 255) / 256, 256, 0, stream>>>(h1, colb, T_ENC, tcol);
        wpack_k3_kernel<<<(twp + 255) / 256, 256, 0, stream>>>(dp_c2_w, wpack, twp);
        gemm(stream, colb, 3 * DMODEL, wpack, DMODEL, h2, DMODEL,
             Menc, DMODEL, 3 * DMODEL, EPI_RELU_BN, dp_c2_b, dp_g2, dp_b2);
        // conv3 (1x1, O=1) + softplus
        dur_kernel<<<Menc, 32, 0, stream>>>(h2, dp_c3_w, dp_c3_b, dur);
    }

    // ---- length regulator ----
    cumsum_kernel<<<BATCH, 32, 0, stream>>>(dur, ends);
    {
        int t1 = BATCH * T_DEC;
        expand_idx_kernel<<<(t1 + 255) / 256, 256, 0, stream>>>(ends, idx, maskf);
        int t2 = BATCH * T_DEC * DMODEL;
        float* dec_in = (cur == x_b) ? x_a : x_b;
        gather_kernel<<<(t2 + 255) / 256, 256, 0, stream>>>(cur, idx, maskf, dec_in, t2);
        cur = dec_in; nxt = (cur == x_a) ? x_b : x_a;
    }

    // ---- decoder ----
    for (int i = 0; i < NLAYER; ++i) {
        mamba_layer(stream, T_DEC, cur, nxt, xz, xi, delta, ygate, xdbl,
                    dec_base[0] + i * ST[0], dec_base[1] + i * ST[1], dec_base[2] + i * ST[2],
                    dec_base[3] + i * ST[3], dec_base[4] + i * ST[4], dec_base[5] + i * ST[5],
                    dec_base[6] + i * ST[6], dec_base[7] + i * ST[7], dec_base[8] + i * ST[8]);
        float* t = cur; cur = nxt; nxt = t;
    }

    // ---- rmsnorm + mel projection (masked) ----
    rmsnorm_kernel<<<BATCH * T_DEC, 32, 0, stream>>>(cur, norm_g, nxt);
    float* mel = (float*)d_out;
    gemm(stream, nxt, DMODEL, out_w_fin, NMELS, mel, NMELS,
         BATCH * T_DEC, NMELS, DMODEL, EPI_ROWSCALE, maskf);
    {
        int t1 = BATCH * T_DEC;
        mask_out_kernel<<<(t1 + 255) / 256, 256, 0, stream>>>(maskf, mel + (size_t)BATCH * T_DEC * NMELS);
    }
    (void)in_sizes; (void)n_in; (void)out_size; (void)ws_size;
}
